// PitchTransformer_89550068121657
// MI455X (gfx1250) — matrix operand order repaired and checked by reading, not compiled, not run on hardware
//
#include <hip/hip_runtime.h>
#include <math.h>

// ---------------- model constants ----------------
#define D_MODEL 1024
#define N_HEADS 16
#define N_KV 8
#define HEAD_DIM 64
#define INTER 4096
#define N_LAYERS 4
#define TEXT_DIM 512
#define OUT_DIM 3
#define NB 2
#define TT 16
#define PP 32
#define FQ 32
#define LL 64
#define SEQ (LL + TT*(PP+1) + TT*FQ)   // 1104
#define ROWS (NB*SEQ)                  // 2208
#define KV_DIM (N_KV*HEAD_DIM)         // 512
#define SCALE 0.125f                   // 64^-0.5

typedef __bf16 bf16;
typedef bf16  v16bf  __attribute__((ext_vector_type(16)));
typedef bf16  bf16x8 __attribute__((ext_vector_type(8)));
typedef float v8f    __attribute__((ext_vector_type(8)));
typedef float v4f    __attribute__((ext_vector_type(4)));
typedef unsigned int u32x4 __attribute__((ext_vector_type(4)));
typedef int i32x8 __attribute__((ext_vector_type(8)));
typedef int i32x4 __attribute__((ext_vector_type(4)));

#if defined(__has_builtin)
#if __has_builtin(__builtin_amdgcn_tensor_load_to_lds) && \
    __has_builtin(__builtin_amdgcn_s_wait_tensorcnt)
#define HAVE_TDM 1
#endif
#endif
#ifndef HAVE_TDM
#define HAVE_TDM 0
#endif

static __device__ __forceinline__ v8f v8f_zero() {
  v8f z;
#pragma unroll
  for (int i = 0; i < 8; ++i) z[i] = 0.0f;
  return z;
}

static __device__ __forceinline__ v16bf combine16(bf16x8 lo, bf16x8 hi) {
  v16bf r;
#pragma unroll
  for (int i = 0; i < 8; ++i) { r[i] = lo[i]; r[i + 8] = hi[i]; }
  return r;
}

static __device__ __forceinline__ v8f wmma_bf16(v16bf a, v16bf b, v8f c) {
  // D(f32,16x16) = A(16x32 bf16) * B(32x16 bf16) + C
  return __builtin_amdgcn_wmma_f32_16x16x32_bf16(false, a, false, b,
                                                 (short)0, c, false, false);
}

// ---------------- embedding ----------------
static __device__ __forceinline__ float pos_enc(float fq, float tm, int d) {
  // half = 512, REF_FREQ = 440, REF_TIME = 1, SIGMA = 1
  float ha = (float)(d >> 1) * (1.0f / 512.0f);
  float fr = fq * (1.0f / 440.0f);
  float tr = tm;
  float phase = 6.283185307179586f * (fr * ha + tr);
  float diff = ha - tr;
  float amp = __expf(-diff * diff) * 0.3989422804014327f;
  return ((d & 1) ? __sinf(phase) : __cosf(phase)) * amp;
}

__global__ void embed_kernel(
    const float* __restrict__ pitch_spec, const float* __restrict__ pitchs,
    const float* __restrict__ pitch_centre, const float* __restrict__ freq_spec,
    const float* __restrict__ freqs, const float* __restrict__ freq_centre,
    const float* __restrict__ text_emb, const float* __restrict__ pitch_W,
    const float* __restrict__ pitch_b, const float* __restrict__ freq_W,
    const float* __restrict__ freq_b, const float* __restrict__ pitchless,
    const float* __restrict__ pitch_tok, const float* __restrict__ freq_tok,
    const float* __restrict__ text_W, const float* __restrict__ text_b,
    float* __restrict__ h)
{
  int row = blockIdx.x;             // 0..ROWS-1
  int b = row / SEQ;
  int s = row - b * SEQ;
  int tid = threadIdx.x;            // 256 threads, 4 dims each
  float out[4];

  if (s < LL) {
    const float* te = text_emb + (size_t)(b * LL + s) * TEXT_DIM;
#pragma unroll
    for (int j = 0; j < 4; ++j) {
      int d = tid * 4 + j;
      const float* wrow = text_W + (size_t)d * TEXT_DIM;
      float acc = text_b[d];
      for (int c = 0; c < TEXT_DIM; ++c) acc += te[c] * wrow[c];
      out[j] = acc;
    }
  } else if (s < LL + TT * (PP + 1)) {
    int idx = s - LL;
    int t = idx / (PP + 1);
    int p = idx - t * (PP + 1);
#pragma unroll
    for (int j = 0; j < 4; ++j) {
      int d = tid * 4 + j;
      float v;
      if (p < PP) {
        float pe = pos_enc(pitchs[p], pitch_centre[t], d);
        v = pitch_spec[(size_t)(b * TT + t) * PP + p] * pitch_W[d] + pitch_b[d] + pe;
      } else {
        v = pitchless[d];
      }
      out[j] = v + pitch_tok[d];
    }
  } else {
    int idx = s - LL - TT * (PP + 1);
    int t = idx / FQ;
    int f = idx - t * FQ;
#pragma unroll
    for (int j = 0; j < 4; ++j) {
      int d = tid * 4 + j;
      float pe = pos_enc(freqs[f], freq_centre[t], d);
      out[j] = freq_spec[(size_t)(b * TT + t) * FQ + f] * freq_W[d] + freq_b[d]
               + pe + freq_tok[d];
    }
  }
#pragma unroll
  for (int j = 0; j < 4; ++j)
    h[(size_t)row * D_MODEL + tid * 4 + j] = out[j];
}

// ---------------- rmsnorm -> bf16 ----------------
__global__ void rmsnorm_kernel(const float* __restrict__ h,
                               const float* __restrict__ w,
                               bf16* __restrict__ xn)
{
  int row = blockIdx.x;
  int tid = threadIdx.x;
  const float* x = h + (size_t)row * D_MODEL;
  float ss = 0.0f;
#pragma unroll
  for (int j = 0; j < 4; ++j) {
    float v = x[tid * 4 + j];
    ss += v * v;
  }
  __shared__ float red[256];
  red[tid] = ss;
  __syncthreads();
  for (int off = 128; off > 0; off >>= 1) {
    if (tid < off) red[tid] += red[tid + off];
    __syncthreads();
  }
  float inv = rsqrtf(red[0] * (1.0f / D_MODEL) + 1e-6f);
#pragma unroll
  for (int j = 0; j < 4; ++j) {
    int d = tid * 4 + j;
    xn[(size_t)row * D_MODEL + d] = (bf16)(w[d] * x[d] * inv);
  }
}

// ---------------- GEMM staging helpers ----------------
static __device__ __forceinline__ void stage_w_tile(
    const float* __restrict__ W, bf16 (*Wt)[32], int nblk, int K,
    int k0, int lrow, int lhalf)
{
  const float* wp = W + (size_t)(nblk + lrow) * K + k0 + lhalf;
  v4f w0 = *(const v4f*)(wp);
  v4f w1 = *(const v4f*)(wp + 4);
  v4f w2 = *(const v4f*)(wp + 8);
  v4f w3 = *(const v4f*)(wp + 12);
  bf16x8 wb0, wb1;
#pragma unroll
  for (int i = 0; i < 4; ++i) {
    wb0[i]     = (bf16)w0[i];
    wb0[i + 4] = (bf16)w1[i];
    wb1[i]     = (bf16)w2[i];
    wb1[i + 4] = (bf16)w3[i];
  }
  *(bf16x8*)&Wt[lrow][lhalf]     = wb0;
  *(bf16x8*)&Wt[lrow][lhalf + 8] = wb1;
}

static __device__ __forceinline__ void stage_a_tile(
    const bf16* __restrict__ A, bf16 (*At)[32], int mblk, int M, int K,
    int k0, int lrow, int lhalf)
{
  int arow = mblk + lrow;
  if (arow >= M) arow = M - 1;     // clamped garbage; stores are guarded
  const bf16* ap = A + (size_t)arow * K + k0 + lhalf;
  *(bf16x8*)&At[lrow][lhalf]     = *(const bf16x8*)(ap);
  *(bf16x8*)&At[lrow][lhalf + 8] = *(const bf16x8*)(ap + 8);
}

#if HAVE_TDM
// Tensor Data Mover: DMA a 128x32 bf16 tile of A[M,K] into LDS.
// D# built per CDNA5 ISA 8.3/8.4 (group0: count/lds/global/type,
// group1: data_size=2B, tensor dims, tile 32x128, row stride K).
// clang-23 toolchain: 6-arg form (extra i32x8 group before cpol).
static __device__ __forceinline__ void tdm_load_a_tile(
    const bf16* gsrc, unsigned lds_addr, int rows_rem, int K)
{
  unsigned long long ga = (unsigned long long)(size_t)gsrc;
  u32x4 g0;
  g0[0] = 1u;                                          // count=1, user mode
  g0[1] = lds_addr;                                    // LDS byte address
  g0[2] = (unsigned)ga;                                // global_addr[31:0]
  g0[3] = (unsigned)((ga >> 32) & 0x1FFFFFFu) | (2u << 30);  // addr[56:32]|type=2
  unsigned td0 = (unsigned)K;          // tensor_dim0 (elements)
  unsigned td1 = (unsigned)rows_rem;   // tensor_dim1 (rows remaining -> OOB=0)
  i32x8 g1;
  g1[0] = 0x00010000;                                  // wg_mask=0, data_size=2B
  g1[1] = (int)((td0 & 0xFFFFu) << 16);                // [63:48]=td0.lo16
  g1[2] = (int)((td0 >> 16) | ((td1 & 0xFFFFu) << 16));// td0.hi16 | td1.lo16
  g1[3] = (int)((td1 >> 16) | (32u << 16));            // td1.hi16 | tile_dim0=32
  g1[4] = 128;                                         // tile_dim1=128, dim2=0
  g1[5] = (int)td0;                                    // tensor_dim0_stride lo32
  g1[6] = 0;                                           // stride hi, dim1_stride
  g1[7] = 0;
  i32x4 g2; g2[0] = 0; g2[1] = 0; g2[2] = 0; g2[3] = 0;
  i32x4 g3; g3[0] = 0; g3[1] = 0; g3[2] = 0; g3[3] = 0;
  i32x8 g4;
#pragma unroll
  for (int i = 0; i < 8; ++i) g4[i] = 0;
  __builtin_amdgcn_tensor_load_to_lds(g0, g1, g2, g3, g4, 0);
}
#endif

// ---------------- generic WMMA GEMM: C = A(bf16)[M,K] * W(f32,[N,K])^T -----
#define EPI_F32   0
#define EPI_GEGLU 1

__global__ __launch_bounds__(256, 2) void gemm_kernel(
    const bf16* __restrict__ A, const float* __restrict__ W,
    const float* __restrict__ bias, const float* __restrict__ residual,
    const float* __restrict__ gate, float* __restrict__ outf,
    bf16* __restrict__ outb, int M, int N, int K, int mode)
{
  __shared__ bf16 At[2][128][32];   // double-buffered
  __shared__ bf16 Wt[2][128][32];

  int tid  = threadIdx.x;
  int lane = tid & 31;
  int wave = tid >> 5;        // 0..7
  int wm   = wave >> 1;       // 0..3 -> 32 rows each
  int wn   = wave & 1;        // 0..1 -> 64 cols each
  int mblk = blockIdx.y * 128;
  int nblk = blockIdx.x * 128;

  v8f c[2][4];
#pragma unroll
  for (int mt = 0; mt < 2; ++mt)
#pragma unroll
    for (int nt = 0; nt < 4; ++nt) c[mt][nt] = v8f_zero();

  int lrow  = tid >> 1;           // 0..127
  int lhalf = (tid & 1) * 16;     // 0 or 16
  int m_    = lane & 15;
  int lgrp  = lane >> 4;
  int ko8   = lgrp * 8;
  int ko16  = lgrp * 16;

  const int T = K >> 5;           // K/32 k-tiles
#if HAVE_TDM
  unsigned lds_at = (unsigned)(size_t)(&At[0][0][0]);
  const bf16* a_org = A + (size_t)mblk * K;
  int rows_rem = M - mblk;
#endif

  // ---- prologue: stage tile 0 into buffer 0 ----
#if HAVE_TDM
  if (wave == 0) tdm_load_a_tile(a_org, lds_at, rows_rem, K);
#else
  stage_a_tile(A, At[0], mblk, M, K, 0, lrow, lhalf);
#endif
  stage_w_tile(W, Wt[0], nblk, K, 0, lrow, lhalf);
#if HAVE_TDM
  if (wave == 0) __builtin_amdgcn_s_wait_tensorcnt(0);
#endif
  __syncthreads();

  for (int i = 0; i < T; ++i) {
    int cb = i & 1;
    int nb = cb ^ 1;
    // ---- stage tile i+1 into the other buffer (overlaps WMMA below) ----
    if (i + 1 < T) {
      int k0n = (i + 1) * 32;
#if HAVE_TDM
      if (wave == 0)
        tdm_load_a_tile(a_org + k0n, lds_at + (unsigned)nb * (128 * 32 * 2),
                        rows_rem, K);
#else
      stage_a_tile(A, At[nb], mblk, M, K, k0n, lrow, lhalf);
#endif
      stage_w_tile(W, Wt[nb], nblk, K, k0n, lrow, lhalf);
      if (i + 2 < T)
        __builtin_prefetch(W + (size_t)(nblk + lrow) * K + (i + 2) * 32 + lhalf,
                           0, 1);
    }

    // ---- compute on current buffer ----
    v16bf af[2];
#pragma unroll
    for (int mt = 0; mt < 2; ++mt) {
      int r = wm * 32 + mt * 16 + m_;
      bf16x8 lo = *(const bf16x8*)&At[cb][r][ko8];
      bf16x8 hi = *(const bf16x8*)&At[cb][r][16 + ko8];
      af[mt] = combine16(lo, hi);
    }
    v16bf bfr[4];
#pragma unroll
    for (int nt = 0; nt < 4; ++nt) {
      int r = wn * 64 + nt * 16 + m_;
      bf16x8 lo = *(const bf16x8*)&Wt[cb][r][ko8];
      bf16x8 hi = *(const bf16x8*)&Wt[cb][r][16 + ko8];
      bfr[nt] = combine16(lo, hi);
    }
#pragma unroll
    for (int mt = 0; mt < 2; ++mt)
#pragma unroll
      for (int nt = 0; nt < 4; ++nt)
        c[mt][nt] = wmma_bf16(af[mt], bfr[nt], c[mt][nt]);

#if HAVE_TDM
    if (wave == 0 && i + 1 < T) __builtin_amdgcn_s_wait_tensorcnt(0);
#endif
    __syncthreads();   // next-buffer writes visible; current buffer reusable
  }

  // epilogue (C layout: lane column n, VGPR r -> row (lgrp*8 + r))
  int nloc = lane & 15;
  int mb   = lgrp * 8;
#pragma unroll
  for (int mt = 0; mt < 2; ++mt) {
#pragma unroll
    for (int nt = 0; nt < 4; ++nt) {
      int gcol = nblk + wn * 64 + nt * 16 + nloc;
      float bv = bias ? bias[gcol] : 0.0f;
#pragma unroll
      for (int r = 0; r < 8; ++r) {
        int grow = mblk + wm * 32 + mt * 16 + mb + r;
        if (grow < M) {
          float acc = c[mt][nt][r] + bv;
          size_t idx = (size_t)grow * N + gcol;
          if (mode == EPI_F32) {
            if (residual) acc += residual[idx];
            outf[idx] = acc;
          } else {  // fused GeGLU: out = bf16( gelu(gate) * acc )
            float gv = gate[idx];
            float ge = 0.5f * gv * (1.0f + erff(gv * 0.7071067811865476f));
            outb[idx] = (bf16)(ge * acc);
          }
        }
      }
    }
  }
}

// ---------------- flash-style WMMA attention -------------------------------
// one wave per (b, head, 16-query tile); writes O as bf16
__global__ __launch_bounds__(32) void attn_kernel(
    const float* __restrict__ qb, const float* __restrict__ kb,
    const float* __restrict__ vb, bf16* __restrict__ ob)
{
  __shared__ bf16 lds_p[16][32];     // P tile re-striped D-layout -> A-layout
  __shared__ bf16 lds_vt[64][33];    // V^T (padded) for contiguous B-frag reads

  int lane = threadIdx.x;
  int qt = blockIdx.x;               // 0..68  (1104/16)
  int hh = blockIdx.y;               // head
  int b  = blockIdx.z;
  int q0 = qt * 16;
  int m_   = lane & 15;
  int lgrp = lane >> 4;
  int ko8  = lgrp * 8;
  int ko16 = lgrp * 16;
  int mb   = lgrp * 8;
  int kvh  = hh >> 1;                // GQA: head h -> kv head h/2

  // Q fragments (16x64 split into two 16x32 chunks), f32 -> bf16
  v16bf qf[2];
#pragma unroll
  for (int cch = 0; cch < 2; ++cch) {
    const float* qrow =
        qb + (size_t)(b * SEQ + q0 + m_) * D_MODEL + hh * HEAD_DIM + cch * 32;
    v16bf f;
#pragma unroll
    for (int i = 0; i < 8; ++i) f[i] = (bf16)qrow[ko8 + i];
#pragma unroll
    for (int i = 0; i < 8; ++i) f[8 + i] = (bf16)qrow[16 + ko8 + i];
    qf[cch] = f;
  }

  float mrow[8], lrow[8];
#pragma unroll
  for (int r = 0; r < 8; ++r) { mrow[r] = -1e30f; lrow[r] = 0.0f; }
  v8f o[4];
#pragma unroll
  for (int nt = 0; nt < 4; ++nt) o[nt] = v8f_zero();

  const int NTILES = (SEQ + 31) / 32;  // 35 (last has 16 valid keys)
  for (int kt = 0; kt < NTILES; ++kt) {
    v8f s0 = v8f_zero(), s1 = v8f_zero();
    // S = Q * K^T : two key n-tiles, K-dim = 64 -> two 32 chunks
#pragma unroll
    for (int j = 0; j < 2; ++j) {
      int key = kt * 32 + j * 16 + m_;
      int keyc = key < SEQ ? key : SEQ - 1;
      const float* krow =
          kb + (size_t)(b * SEQ + keyc) * KV_DIM + kvh * HEAD_DIM;
#pragma unroll
      for (int cch = 0; cch < 2; ++cch) {
        v16bf f;
#pragma unroll
        for (int i = 0; i < 16; ++i) f[i] = (bf16)krow[cch * 32 + (i < 8 ? ko8 + i : 8 + ko8 + i)];
        if (j == 0) s0 = wmma_bf16(qf[cch], f, s0);
        else        s1 = wmma_bf16(qf[cch], f, s1);
      }
    }
    // scale, mask, online softmax (16-lane shuffle reductions per row)
    int key0 = kt * 32 + m_;
    int key1 = key0 + 16;
    float msk0 = key0 < SEQ ? 0.0f : -1e30f;
    float msk1 = key1 < SEQ ? 0.0f : -1e30f;
    float p0[8], p1[8], alpha[8];
#pragma unroll
    for (int r = 0; r < 8; ++r) {
      float a0 = s0[r] * SCALE + msk0;
      float a1 = s1[r] * SCALE + msk1;
      float tm = fmaxf(a0, a1);
#pragma unroll
      for (int off = 1; off < 16; off <<= 1)
        tm = fmaxf(tm, __shfl_xor(tm, off, 32));
      float nm = fmaxf(mrow[r], tm);
      float al = __expf(mrow[r] - nm);
      float e0 = __expf(a0 - nm);
      float e1 = __expf(a1 - nm);
      float rs = e0 + e1;
#pragma unroll
      for (int off = 1; off < 16; off <<= 1)
        rs += __shfl_xor(rs, off, 32);
      lrow[r] = lrow[r] * al + rs;
      mrow[r] = nm;
      p0[r] = e0; p1[r] = e1; alpha[r] = al;
    }
#pragma unroll
    for (int nt = 0; nt < 4; ++nt)
#pragma unroll
      for (int r = 0; r < 8; ++r) o[nt][r] = o[nt][r] * alpha[r];

    // P (D layout) -> LDS -> A layout
#pragma unroll
    for (int r = 0; r < 8; ++r) {
      lds_p[mb + r][m_]      = (bf16)p0[r];
      lds_p[mb + r][16 + m_] = (bf16)p1[r];
    }
    // V tile transposed into LDS: lane loads one key row, scatters columns
    {
      int key = kt * 32 + lane;
      int keyc = key < SEQ ? key : SEQ - 1;
      const float* vrow =
          vb + (size_t)(b * SEQ + keyc) * KV_DIM + kvh * HEAD_DIM;
#pragma unroll
      for (int d = 0; d < HEAD_DIM; ++d) lds_vt[d][lane] = (bf16)vrow[d];
    }
    asm volatile("s_wait_dscnt 0" ::: "memory");

    v16bf pa;
#pragma unroll
    for (int i = 0; i < 8; ++i) pa[i] = lds_p[m_][ko8 + i];
#pragma unroll
    for (int i = 0; i < 8; ++i) pa[8 + i] = lds_p[m_][16 + ko8 + i];
#pragma unroll
    for (int nt = 0; nt < 4; ++nt) {
      v16bf vf;
#pragma unroll
      for (int i = 0; i < 16; ++i) vf[i] = lds_vt[nt * 16 + m_][i < 8 ? ko8 + i : 8 + ko8 + i];
      o[nt] = wmma_bf16(pa, vf, o[nt]);
    }
    asm volatile("s_wait_dscnt 0" ::: "memory");
  }

#pragma unroll
  for (int r = 0; r < 8; ++r) {
    float inv = 1.0f / lrow[r];
    int grow = b * SEQ + q0 + mb + r;
#pragma unroll
    for (int nt = 0; nt < 4; ++nt)
      ob[(size_t)grow * D_MODEL + hh * HEAD_DIM + nt * 16 + m_] =
          (bf16)(o[nt][r] * inv);
  }
}

// ---------------- classifier ----------------
__global__ void cls_kernel(const float* __restrict__ h,
                           const float* __restrict__ cw,
                           const float* __restrict__ cb,
                           float* __restrict__ out)
{
  int idx = blockIdx.x;  // 0..NB*TT*(PP+1)-1
  int b = idx / (TT * (PP + 1));
  int tp = idx - b * (TT * (PP + 1));
  int s = LL + tp;
  const float* x = h + (size_t)(b * SEQ + s) * D_MODEL;
  int tid = threadIdx.x;  // 128
  float acc[3] = {0.0f, 0.0f, 0.0f};
  for (int d = tid; d < D_MODEL; d += 128) {
    float xv = x[d];
#pragma unroll
    for (int o3 = 0; o3 < 3; ++o3) acc[o3] += xv * cw[o3 * D_MODEL + d];
  }
  __shared__ float red[3][128];
#pragma unroll
  for (int o3 = 0; o3 < 3; ++o3) red[o3][tid] = acc[o3];
  __syncthreads();
  for (int off = 64; off > 0; off >>= 1) {
    if (tid < off)
#pragma unroll
      for (int o3 = 0; o3 < 3; ++o3) red[o3][tid] += red[o3][tid + off];
    __syncthreads();
  }
  if (tid < 3) out[(size_t)idx * 3 + tid] = red[tid][0] + cb[tid];
}

// ---------------- host orchestration ----------------
extern "C" void kernel_launch(void* const* d_in, const int* in_sizes, int n_in,
                              void* d_out, int out_size, void* d_ws,
                              size_t ws_size, hipStream_t stream)
{
  (void)in_sizes; (void)n_in; (void)out_size; (void)ws_size;
  // setup_inputs() flat order (nested dicts depth-first, insertion order)
  const float* pitch_spec   = (const float*)d_in[0];
  const float* pitchs       = (const float*)d_in[1];
  const float* pitch_centre = (const float*)d_in[2];
  const float* freq_spec    = (const float*)d_in[3];
  const float* freqs        = (const float*)d_in[4];
  const float* freq_centre  = (const float*)d_in[5];
  const float* text_emb     = (const float*)d_in[6];
  const float* pitch_W      = (const float*)d_in[7];
  const float* pitch_b      = (const float*)d_in[8];
  const float* freq_W       = (const float*)d_in[9];
  const float* freq_b       = (const float*)d_in[10];
  const float* pitchless    = (const float*)d_in[11];
  const float* pitch_tok    = (const float*)d_in[12];
  const float* freq_tok     = (const float*)d_in[13];
  const float* text_W       = (const float*)d_in[14];
  const float* text_b       = (const float*)d_in[15];
  const float* cls_W        = (const float*)d_in[16];
  const float* cls_b        = (const float*)d_in[17];
  const float* Wq_all = (const float*)d_in[18];
  const float* bq_all = (const float*)d_in[19];
  const float* Wk_all = (const float*)d_in[20];
  const float* bk_all = (const float*)d_in[21];
  const float* Wv_all = (const float*)d_in[22];
  const float* bv_all = (const float*)d_in[23];
  const float* Wo_all = (const float*)d_in[24];
  const float* Wg_all = (const float*)d_in[25];
  const float* Wu_all = (const float*)d_in[26];
  const float* Wd_all = (const float*)d_in[27];
  const float* ln1_all = (const float*)d_in[28];
  const float* ln2_all = (const float*)d_in[29];

  // workspace carve-out (256B aligned slices)
  char* wsp = (char*)d_ws;
  auto carve = [&wsp](size_t bytes) -> void* {
    void* p = (void*)wsp;
    wsp += (bytes + 255) & ~(size_t)255;
    return p;
  };
  float* h_buf  = (float*)carve((size_t)ROWS * D_MODEL * 4);  // hidden f32
  bf16*  xn_buf = (bf16*) carve((size_t)ROWS * D_MODEL * 2);  // normed bf16
  float* q_buf  = (float*)carve((size_t)ROWS * D_MODEL * 4);
  float* k_buf  = (float*)carve((size_t)ROWS * KV_DIM * 4);
  float* v_buf  = (float*)carve((size_t)ROWS * KV_DIM * 4);
  bf16*  o_buf  = (bf16*) carve((size_t)ROWS * D_MODEL * 2);  // attn out bf16
  float* g_buf  = (float*)carve((size_t)ROWS * INTER * 4);    // gate f32
  bf16*  a_buf  = (bf16*) carve((size_t)ROWS * INTER * 2);    // geglu bf16

  const int MBLK = (ROWS + 127) / 128;  // 18

  embed_kernel<<<ROWS, 256, 0, stream>>>(
      pitch_spec, pitchs, pitch_centre, freq_spec, freqs, freq_centre,
      text_emb, pitch_W, pitch_b, freq_W, freq_b, pitchless, pitch_tok,
      freq_tok, text_W, text_b, h_buf);

  for (int l = 0; l < N_LAYERS; ++l) {
    const float* Wq = Wq_all + (size_t)l * D_MODEL * D_MODEL;
    const float* bq = bq_all + (size_t)l * D_MODEL;
    const float* Wk = Wk_all + (size_t)l * KV_DIM * D_MODEL;
    const float* bk = bk_all + (size_t)l * KV_DIM;
    const float* Wv = Wv_all + (size_t)l * KV_DIM * D_MODEL;
    const float* bv = bv_all + (size_t)l * KV_DIM;
    const float* Wo = Wo_all + (size_t)l * D_MODEL * D_MODEL;
    const float* Wg = Wg_all + (size_t)l * INTER * D_MODEL;
    const float* Wu = Wu_all + (size_t)l * INTER * D_MODEL;
    const float* Wd = Wd_all + (size_t)l * D_MODEL * INTER;
    const float* ln1 = ln1_all + (size_t)l * D_MODEL;
    const float* ln2 = ln2_all + (size_t)l * D_MODEL;

    rmsnorm_kernel<<<ROWS, 256, 0, stream>>>(h_buf, ln1, xn_buf);

    gemm_kernel<<<dim3(D_MODEL / 128, MBLK), 256, 0, stream>>>(
        xn_buf, Wq, bq, nullptr, nullptr, q_buf, nullptr,
        ROWS, D_MODEL, D_MODEL, EPI_F32);
    gemm_kernel<<<dim3(KV_DIM / 128, MBLK), 256, 0, stream>>>(
        xn_buf, Wk, bk, nullptr, nullptr, k_buf, nullptr,
        ROWS, KV_DIM, D_MODEL, EPI_F32);
    gemm_kernel<<<dim3(KV_DIM / 128, MBLK), 256, 0, stream>>>(
        xn_buf, Wv, bv, nullptr, nullptr, v_buf, nullptr,
        ROWS, KV_DIM, D_MODEL, EPI_F32);

    attn_kernel<<<dim3(SEQ / 16, N_HEADS, NB), 32, 0, stream>>>(
        q_buf, k_buf, v_buf, o_buf);

    // h = h + o @ Wo^T   (residual fused)
    gemm_kernel<<<dim3(D_MODEL / 128, MBLK), 256, 0, stream>>>(
        o_buf, Wo, nullptr, h_buf, nullptr, h_buf, nullptr,
        ROWS, D_MODEL, D_MODEL, EPI_F32);

    rmsnorm_kernel<<<ROWS, 256, 0, stream>>>(h_buf, ln2, xn_buf);

    // g = x @ Wg^T ; a = bf16(gelu(g) * (x @ Wu^T))  (fused GeGLU epilogue)
    gemm_kernel<<<dim3(INTER / 128, MBLK), 256, 0, stream>>>(
        xn_buf, Wg, nullptr, nullptr, nullptr, g_buf, nullptr,
        ROWS, INTER, D_MODEL, EPI_F32);
    gemm_kernel<<<dim3(INTER / 128, MBLK), 256, 0, stream>>>(
        xn_buf, Wu, nullptr, nullptr, g_buf, nullptr, a_buf,
        ROWS, INTER, D_MODEL, EPI_GEGLU);

    // h = h + a @ Wd^T
    gemm_kernel<<<dim3(D_MODEL / 128, MBLK), 256, 0, stream>>>(
        a_buf, Wd, nullptr, h_buf, nullptr, h_buf, nullptr,
        ROWS, D_MODEL, INTER, EPI_F32);
  }

  cls_kernel<<<NB * TT * (PP + 1), 128, 0, stream>>>(
      h_buf, cls_W, cls_b, (float*)d_out);
}
